// CantorAttention_61400852464246
// MI455X (gfx1250) — compile-verified, matrix operand order checked by reading, not run on hardware
//
#include <hip/hip_runtime.h>
#include <hip/hip_bf16.h>

// Problem constants (fixed by the reference)
#define B_    2
#define S_    2048
#define DIM_  1024
#define H_    16
#define HD_   64
#define KN_   64
#define SCALE_ 0.125f   // 1/sqrt(64)

typedef __attribute__((ext_vector_type(16))) __bf16 v16bf;
typedef __attribute__((ext_vector_type(8)))  float  v8f;

// async-copy builtin wants pointers to int4 (vector_size(16)) in AS1/AS3
typedef int v4i __attribute__((vector_size(16)));
typedef __attribute__((address_space(1))) v4i as1_v4i;
typedef __attribute__((address_space(3))) v4i as3_v4i;

struct Raw32 { uint4 lo; uint4 hi; };  // 32 bytes == v16bf

#if __has_builtin(__builtin_amdgcn_global_load_async_to_lds_b128)
#define HAVE_ASYNC_LDS 1
#else
#define HAVE_ASYNC_LDS 0
#endif

#if HAVE_ASYNC_LDS
#if __has_builtin(__builtin_amdgcn_s_wait_asynccnt)
#define WAIT_ASYNC(n) __builtin_amdgcn_s_wait_asynccnt(n)
#else
#define WAIT_ASYNC(n) asm volatile("s_wait_asynccnt %0" :: "i"(n) : "memory")
#endif
#endif

// ---------------------------------------------------------------------------
// f32 -> bf16 (round-to-nearest-even), stored as raw u16
// ---------------------------------------------------------------------------
__device__ __forceinline__ unsigned short f32_to_bf16u(float f) {
    union { float f; unsigned u; } a; a.f = f;
    unsigned r = a.u + 0x7FFFu + ((a.u >> 16) & 1u);
    return (unsigned short)(r >> 16);
}

__global__ __launch_bounds__(256) void cvt_bf16_kernel(const float* __restrict__ in,
                                                       unsigned short* __restrict__ out,
                                                       int n) {
    int i = blockIdx.x * 256 + threadIdx.x;
    if (i < n) out[i] = f32_to_bf16u(in[i]);
}

// ---------------------------------------------------------------------------
// Load one 16x32 bf16 WMMA fragment slice for this lane.
// Lane l<16 supplies K {0..7,16..23}; lane l>=16 supplies K {8..15,24..31}:
// two contiguous 16B chunks at +0 and +16 elements (lane-half offset folded
// into the pointer by the caller). Works for global or LDS (generic) memory.
// ---------------------------------------------------------------------------
__device__ __forceinline__ v16bf load_frag16(const unsigned short* p) {
    Raw32 r;
    r.lo = *reinterpret_cast<const uint4*>(p);
    r.hi = *reinterpret_cast<const uint4*>(p + 16);
    return __builtin_bit_cast(v16bf, r);
}

// ---------------------------------------------------------------------------
// WMMA GEMM:  C[M,N] = A[M,K] * W[N,K]^T + bias[N]
// Workgroup = 128 threads (4 waves). Tile: M=128 x N=64, K staged by 64.
// The shared 64x64-bf16 B tile is double-buffered in LDS, filled with
// GLOBAL_LOAD_ASYNC_TO_LDS_B128 (ASYNCcnt) when available.
// Each wave: rows [mWave, mWave+32) x all 64 cols -> 2x4 v8f accumulators,
// 16 WMMAs per K-stage.
// MODE 0: store row-major C[m*N+n]  (final out-projection -> d_out)
// MODE 1: scatter into qkv buffer laid out [3][B][H][S][64]
// ---------------------------------------------------------------------------
template <int MODE>
__global__ __launch_bounds__(128) void wmma_gemm_kernel(
    const unsigned short* __restrict__ A,
    const unsigned short* __restrict__ W,
    const float* __restrict__ bias,
    float* __restrict__ C,
    int M, int N, int K)
{
    __shared__ unsigned short Bt[2][64 * 64];   // 2 x 8KB

    const int tid    = threadIdx.x;     // 0..127
    const int wave   = tid >> 5;
    const int lane   = tid & 31;
    const int lane15 = lane & 15;
    const int half   = lane >> 4;       // 0: K 0..7/16..23, 1: K 8..15/24..31
    const int off    = half * 8;

    const int mTile = blockIdx.y * 128;
    const int nTile = blockIdx.x * 64;
    const int mWave = mTile + wave * 32;

    v8f acc[2][4] = { { {}, {}, {}, {} }, { {}, {}, {}, {} } };

    const unsigned short* aRow0 = A + (size_t)(mWave + lane15) * K + off;
    const unsigned short* aRow1 = A + (size_t)(mWave + 16 + lane15) * K + off;

    // ---- cooperative B-tile stage: 64 rows x 64 k (bf16) = 512 16B chunks ----
    auto issue_stage = [&](int buf, int k0) {
#pragma unroll
        for (int i = 0; i < 4; ++i) {
            const int c   = i * 128 + tid;    // 16B chunk id, 0..511
            const int row = c >> 3;           // 8 chunks per row
            const int cw  = c & 7;
            const unsigned short* g = W + (size_t)(nTile + row) * K + k0 + cw * 8;
            unsigned short*       l = &Bt[buf][row * 64 + cw * 8];
#if HAVE_ASYNC_LDS
            __builtin_amdgcn_global_load_async_to_lds_b128(
                (as1_v4i*)g, (as3_v4i*)l, 0, 0);
#else
            *reinterpret_cast<uint4*>(l) = *reinterpret_cast<const uint4*>(g);
#endif
        }
    };

    const int nStages = K / 64;
    issue_stage(0, 0);

    for (int st = 0; st < nStages; ++st) {
        const int nxt = st + 1;
        if (nxt < nStages) {
            issue_stage(nxt & 1, nxt * 64);
#if HAVE_ASYNC_LDS
            WAIT_ASYNC(4);                    // stage st's 4 copies done (in-order)
#endif
        } else {
#if HAVE_ASYNC_LDS
            WAIT_ASYNC(0);
#endif
        }
        __syncthreads();                      // all waves' copies for buf st&1 visible

        const unsigned short* bt = &Bt[st & 1][0];
        const int k0 = st * 64;
#pragma unroll
        for (int kc = 0; kc < 2; ++kc) {
            v16bf a0 = load_frag16(aRow0 + k0 + kc * 32);
            v16bf a1 = load_frag16(aRow1 + k0 + kc * 32);
#pragma unroll
            for (int t = 0; t < 4; ++t) {
                v16bf b = load_frag16(bt + (t * 16 + lane15) * 64 + kc * 32 + off);
                acc[0][t] = __builtin_amdgcn_wmma_f32_16x16x32_bf16(
                                false, a0, false, b, (short)0, acc[0][t], false, false);
                acc[1][t] = __builtin_amdgcn_wmma_f32_16x16x32_bf16(
                                false, a1, false, b, (short)0, acc[1][t], false, false);
            }
        }
        __syncthreads();                      // done reading buf before it is refilled
    }

    // Epilogue: D layout -> lane n=lane15, VGPR r holds row r + 8*half
#pragma unroll
    for (int af = 0; af < 2; ++af) {
#pragma unroll
        for (int t = 0; t < 4; ++t) {
            const int n  = nTile + t * 16 + lane15;
            const float bn = bias[n];
#pragma unroll
            for (int r = 0; r < 8; ++r) {
                const int m = mWave + af * 16 + r + half * 8;
                const float val = acc[af][t][r] + bn;
                if (MODE == 0) {
                    C[(size_t)m * N + n] = val;
                } else {
                    // n = part*1024 + h*64 + d ; m = b*S + s
                    const int part = n >> 10;
                    const int h    = (n >> 6) & (H_ - 1);
                    const int d    = n & (HD_ - 1);
                    const int b    = m >> 11;          // S = 2048
                    const int s    = m & (S_ - 1);
                    const size_t idx = ((((size_t)part * B_ + b) * H_ + h) * S_ + s) * HD_ + d;
                    C[idx] = val;
                }
            }
        }
    }
}

// ---------------------------------------------------------------------------
// Gathered attention: one wave per (b, h, s).
// qkv layout: [3][B][H][S][64] f32.  Output: [B][S][DIM] f32.
// ---------------------------------------------------------------------------
__global__ __launch_bounds__(32) void attn_kernel(
    const float* __restrict__ qkv,
    const int*   __restrict__ routes,
    float*       __restrict__ attn_out)
{
    const int s = blockIdx.x, h = blockIdx.y, b = blockIdx.z;
    const int tid = threadIdx.x;

    __shared__ float qs[HD_];
    __shared__ float sc[KN_];
    __shared__ int   rt[KN_];

    const size_t headBase = (((size_t)b * H_ + h) * S_) * HD_;
    const size_t planeSz  = (size_t)B_ * H_ * S_ * HD_;
    const float* qp = qkv + headBase + (size_t)s * HD_;
    const float* kb = qkv + planeSz     + headBase;
    const float* vb = qkv + 2 * planeSz + headBase;

    qs[tid]      = qp[tid];
    qs[tid + 32] = qp[tid + 32];
    rt[tid]      = routes[s * KN_ + tid];
    rt[tid + 32] = routes[s * KN_ + tid + 32];
    __syncthreads();

    // scores: each thread handles neighbors tid and tid+32
    float my[2];
#pragma unroll
    for (int jj = 0; jj < 2; ++jj) {
        const int j = tid + jj * 32;
        const float4* kp = reinterpret_cast<const float4*>(kb + (size_t)rt[j] * HD_);
        float acc = 0.f;
#pragma unroll
        for (int i = 0; i < 16; ++i) {
            float4 kv = kp[i];
            acc += kv.x * qs[4*i] + kv.y * qs[4*i+1] + kv.z * qs[4*i+2] + kv.w * qs[4*i+3];
        }
        my[jj] = acc * SCALE_;
    }

    // wave32 softmax reduction
    float mx = fmaxf(my[0], my[1]);
#pragma unroll
    for (int o = 16; o >= 1; o >>= 1) mx = fmaxf(mx, __shfl_xor(mx, o));
    const float e0 = __expf(my[0] - mx);
    const float e1 = __expf(my[1] - mx);
    float sum = e0 + e1;
#pragma unroll
    for (int o = 16; o >= 1; o >>= 1) sum += __shfl_xor(sum, o);
    const float inv = 1.0f / sum;
    sc[tid]      = e0 * inv;
    sc[tid + 32] = e1 * inv;
    __syncthreads();

    // out[d] = sum_j attn[j] * v[route_j][d]; coalesced V reads across lanes
    float o0 = 0.f, o1 = 0.f;
    for (int j = 0; j < KN_; ++j) {
        const float* vp = vb + (size_t)rt[j] * HD_;
        const float w = sc[j];
        o0 += w * vp[tid];
        o1 += w * vp[tid + 32];
    }
    float* op = attn_out + ((size_t)b * S_ + s) * DIM_ + h * HD_;
    op[tid]      = o0;
    op[tid + 32] = o1;
}

// ---------------------------------------------------------------------------
// Orchestration
// ---------------------------------------------------------------------------
extern "C" void kernel_launch(void* const* d_in, const int* in_sizes, int n_in,
                              void* d_out, int out_size, void* d_ws, size_t ws_size,
                              hipStream_t stream) {
    const float* x      = (const float*)d_in[0];
    const int*   routes = (const int*)  d_in[1];
    const float* qkv_w  = (const float*)d_in[2];
    const float* qkv_b  = (const float*)d_in[3];
    const float* out_w  = (const float*)d_in[4];
    const float* out_b  = (const float*)d_in[5];
    float* out = (float*)d_out;

    const int M = B_ * S_;          // 4096
    char* ws = (char*)d_ws;
    size_t off = 0;
    auto take = [&](size_t bytes) -> char* {
        char* p = ws + off;
        off += (bytes + 255) & ~(size_t)255;
        return p;
    };
    unsigned short* x_bf    = (unsigned short*)take((size_t)M * DIM_ * 2);
    unsigned short* qw_bf   = (unsigned short*)take((size_t)3 * DIM_ * DIM_ * 2);
    unsigned short* ow_bf   = (unsigned short*)take((size_t)DIM_ * DIM_ * 2);
    float*          qkv     = (float*)        take((size_t)3 * B_ * H_ * S_ * HD_ * 4);
    float*          aout    = (float*)        take((size_t)M * DIM_ * 4);
    unsigned short* aout_bf = (unsigned short*)take((size_t)M * DIM_ * 2);

    // 1) bf16 conversions
    cvt_bf16_kernel<<<(M * DIM_) / 256, 256, 0, stream>>>(x, x_bf, M * DIM_);
    cvt_bf16_kernel<<<(3 * DIM_ * DIM_) / 256, 256, 0, stream>>>(qkv_w, qw_bf, 3 * DIM_ * DIM_);
    cvt_bf16_kernel<<<(DIM_ * DIM_) / 256, 256, 0, stream>>>(out_w, ow_bf, DIM_ * DIM_);

    // 2) QKV projection GEMM (WMMA bf16 + async LDS staging) -> [3][B][H][S][64]
    wmma_gemm_kernel<1><<<dim3((3 * DIM_) / 64, M / 128), 128, 0, stream>>>(
        x_bf, qw_bf, qkv_b, qkv, M, 3 * DIM_, DIM_);

    // 3) Gathered attention + softmax
    attn_kernel<<<dim3(S_, H_, B_), 32, 0, stream>>>(qkv, routes, aout);

    // 4) Output projection GEMM (WMMA bf16) -> d_out
    cvt_bf16_kernel<<<(M * DIM_) / 256, 256, 0, stream>>>(aout, aout_bf, M * DIM_);
    wmma_gemm_kernel<0><<<dim3(DIM_ / 64, M / 128), 128, 0, stream>>>(
        aout_bf, ow_bf, out_b, out, M, DIM_, DIM_);
}
